// DecoderBlock_19069654794585
// MI455X (gfx1250) — compile-verified, matrix operand order checked by reading, not run on hardware
//
#include <hip/hip_runtime.h>
#include <math.h>

// ---------------------------------------------------------------------------
// DecoderBlock for MI455X (gfx1250, wave32).
// B=4, N=1024, SK=1024, D=1024, H=16, HD=64.
// Self-attention collapses analytically (softmax rows sum to exactly 1) to a
// constant row vector; enc-dec attention streams 8.6GB of K/V (HBM-bound,
// ~0.37ms floor at 23.3 TB/s).  Dense GEMMs: v_wmma_f32_16x16x32_f16.
// K stream staged into LDS via the Tensor Data Mover (double buffered,
// s_wait_tensorcnt), V stream prefetched with global_prefetch_b8.
// ---------------------------------------------------------------------------

typedef __attribute__((ext_vector_type(16))) _Float16 v16h;
typedef __attribute__((ext_vector_type(8)))  float    v8f;
typedef unsigned int v4u __attribute__((ext_vector_type(4)));
typedef int          v8i __attribute__((ext_vector_type(8)));
typedef int          v4i __attribute__((ext_vector_type(4)));

#define BN   4096      // B*N rows
#define DIM  1024
#define SKL  1024
#define HDIM 64

#if defined(__HIP_DEVICE_COMPILE__) && __has_builtin(__builtin_amdgcn_tensor_load_to_lds) && __has_builtin(__builtin_amdgcn_s_wait_tensorcnt)
#define USE_TDM 1
#else
#define USE_TDM 0
#endif

// ---------------------------------------------------------------------------
// WMMA 16x16 output tile of  C = A(MxK,row-major) * B(KxN,row-major), f16
// multiply / f32 accumulate.  Layouts per CDNA5 ISA 7.12.2.
// ---------------------------------------------------------------------------
__device__ __forceinline__ v8f wmma_tile(const float* __restrict__ A, int lda,
                                         const float* __restrict__ B, int ldb,
                                         int m0, int n0, int K) {
  const int lane = threadIdx.x & 31;
  const int half = lane >> 4;
  const int idx  = lane & 15;
  const int kb   = half * 8;
  const int m = m0 + idx;
  const int n = n0 + idx;
  v8f acc = {0.f, 0.f, 0.f, 0.f, 0.f, 0.f, 0.f, 0.f};
  for (int k0 = 0; k0 < K; k0 += 32) {
    const float* arow = A + (size_t)m * lda + k0;
    float4 a0 = *(const float4*)(arow + kb);
    float4 a1 = *(const float4*)(arow + kb + 4);
    float4 a2 = *(const float4*)(arow + 16 + kb);
    float4 a3 = *(const float4*)(arow + 16 + kb + 4);
    v16h av, bv;
    av[0]  = (_Float16)a0.x; av[1]  = (_Float16)a0.y;
    av[2]  = (_Float16)a0.z; av[3]  = (_Float16)a0.w;
    av[4]  = (_Float16)a1.x; av[5]  = (_Float16)a1.y;
    av[6]  = (_Float16)a1.z; av[7]  = (_Float16)a1.w;
    av[8]  = (_Float16)a2.x; av[9]  = (_Float16)a2.y;
    av[10] = (_Float16)a2.z; av[11] = (_Float16)a2.w;
    av[12] = (_Float16)a3.x; av[13] = (_Float16)a3.y;
    av[14] = (_Float16)a3.z; av[15] = (_Float16)a3.w;
    const float* bcol  = B + (size_t)(k0 + kb)      * ldb + n;
    const float* bcol2 = B + (size_t)(k0 + 16 + kb) * ldb + n;
#pragma unroll
    for (int e = 0; e < 8; ++e) bv[e]     = (_Float16)bcol[(size_t)e * ldb];
#pragma unroll
    for (int e = 0; e < 8; ++e) bv[8 + e] = (_Float16)bcol2[(size_t)e * ldb];
    acc = __builtin_amdgcn_wmma_f32_16x16x32_f16(false, av, false, bv,
                                                 (short)0, acc, false, false);
  }
  return acc;
}

// ------------------------- tiny reductions ---------------------------------
__global__ void colsum_kernel(const float* __restrict__ x, float* __restrict__ xsum) {
  int d = blockIdx.x * 256 + threadIdx.x;
  float s = 0.f;
  for (int r = 0; r < BN; ++r) s += x[(size_t)r * DIM + d];
  xsum[d] = s;
}

__global__ void vred_kernel(const float* __restrict__ xsum,
                            const float* __restrict__ qkv_w,
                            const float* __restrict__ qkv_b,
                            float* __restrict__ vred) {
  int f = threadIdx.x;  // 0..63
  float acc = 0.f;
#pragma unroll
  for (int h = 0; h < 16; ++h) acc += 4096.f * qkv_b[2048 + h * 64 + f];
  for (int dd = 0; dd < DIM; ++dd) {
    float xs = xsum[dd];
    const float* wrow = qkv_w + (size_t)dd * 3072 + 2048 + f;
#pragma unroll
    for (int h = 0; h < 16; ++h) acc += xs * wrow[h * 64];
  }
  vred[f] = acc;
}

__global__ void sarow_kernel(const float* __restrict__ vred,
                             const float* __restrict__ w,
                             const float* __restrict__ b,
                             float* __restrict__ sarow) {
  __shared__ float vl[64];
  int t = threadIdx.x;
  if (t < 64) vl[t] = vred[t];
  __syncthreads();
  int d = blockIdx.x * 256 + t;
  float acc = b[d];
  for (int j = 0; j < DIM; ++j) acc += vl[j & 63] * w[(size_t)j * DIM + d];
  sarow[d] = acc;
}

// ----------------------- fused add + LayerNorm -----------------------------
__global__ void add_ln_kernel(const float* __restrict__ a,
                              const float* __restrict__ bsrc, int bcast,
                              const float* __restrict__ g,
                              const float* __restrict__ be,
                              float* __restrict__ out) {
  __shared__ float2 red[256];
  int row = blockIdx.x, t = threadIdx.x;
  const float* ar = a + (size_t)row * DIM;
  const float* br = bcast ? bsrc : (bsrc + (size_t)row * DIM);
  float vals[4];
  float s = 0.f, sq = 0.f;
#pragma unroll
  for (int i = 0; i < 4; ++i) {
    int d = i * 256 + t;
    float v = ar[d] + br[d];
    vals[i] = v; s += v; sq += v * v;
  }
  red[t] = make_float2(s, sq);
  __syncthreads();
  for (int o = 128; o > 0; o >>= 1) {
    if (t < o) { red[t].x += red[t + o].x; red[t].y += red[t + o].y; }
    __syncthreads();
  }
  float mean = red[0].x * (1.f / DIM);
  float var  = red[0].y * (1.f / DIM) - mean * mean;
  float inv  = rsqrtf(var + 1e-6f);
#pragma unroll
  for (int i = 0; i < 4; ++i) {
    int d = i * 256 + t;
    out[(size_t)row * DIM + d] = (vals[i] - mean) * inv * g[d] + be[d];
  }
}

// ----------------------- generic WMMA GEMM + bias (+gelu) ------------------
__global__ void gemm_bias_kernel(const float* __restrict__ A,
                                 const float* __restrict__ B,
                                 const float* __restrict__ bias,
                                 float* __restrict__ C,
                                 int M, int N, int K, int act) {
  int wave   = blockIdx.x * (blockDim.x >> 5) + (threadIdx.x >> 5);
  int ntiles = N >> 4;
  int mt = wave / ntiles, nt = wave - mt * ntiles;
  v8f acc = wmma_tile(A, K, B, N, mt * 16, nt * 16, K);
  int lane = threadIdx.x & 31, half = lane >> 4, idx = lane & 15;
  int n = nt * 16 + idx;
  float bb = bias[n];
#pragma unroll
  for (int v = 0; v < 8; ++v) {
    int m = mt * 16 + half * 8 + v;
    float val = acc[v] + bb;
    if (act) val = 0.5f * val * (1.0f + erff(val * 0.70710678118654752f));
    C[(size_t)m * N + n] = val;
  }
}

// qc GEMM: qc[n,h,b,hd] = (h1 @ q_w + q_b)[b*N+n, h*64+hd]  (scatter epilogue)
__global__ void qc_gemm_kernel(const float* __restrict__ h1,
                               const float* __restrict__ q_w,
                               const float* __restrict__ q_b,
                               float* __restrict__ qc) {
  int wave = blockIdx.x * (blockDim.x >> 5) + (threadIdx.x >> 5);
  int mt = wave >> 6, nt = wave & 63;  // 256 x 64 tiles
  v8f acc = wmma_tile(h1, DIM, q_w, DIM, mt * 16, nt * 16, DIM);
  int lane = threadIdx.x & 31, half = lane >> 4, idx = lane & 15;
  int n = nt * 16 + idx;
  float bb = q_b[n];
  int h = n >> 6, hd = n & 63;
#pragma unroll
  for (int v = 0; v < 8; ++v) {
    int m  = mt * 16 + half * 8 + v;
    int b  = m >> 10;        // batch
    int ns = m & 1023;       // sequence pos
    qc[(((size_t)ns * 16 + h) * 4 + b) * HDIM + hd] = acc[v] + bb;
  }
}

// ----------------------- enc-dec attention (HBM-bound) ---------------------
// one workgroup per (n,h): S = Q[4,64]·K[1024,64]^T, softmax over SK, O = P·V.
// K is staged into LDS by the Tensor Data Mover (double buffered, padded
// rows of 68 floats to dodge LDS bank conflicts); V is streamed coalesced.
#define KROWS 64                 // rows per TDM tile
#define KPAD  68                 // floats per padded LDS row (64 + 4)

#if USE_TDM
__device__ __forceinline__ void tdm_load_k_tile(const float* kb, unsigned lds_off,
                                                int tile) {
  unsigned long long ga =
      (unsigned long long)(uintptr_t)(kb + (size_t)tile * KROWS * HDIM);
  // D# group0: count=1 | lds_addr | global_addr | type=2
  v4u g0 = { 1u, lds_off, (unsigned)ga, (unsigned)(ga >> 32) | (2u << 30) };
  // D# group1: data_size=4B, pad_enable, pad_interval=64DW, pad_amount=4DW,
  // tensor_dim0=64, tensor_dim1=1024, tile_dim0=64, tile_dim1=KROWS,
  // tensor_dim0_stride=64
  v8i g1 = { (int)((2u << 16) | (1u << 20) | (5u << 22) | (3u << 25)),
             (int)(64u << 16),
             (int)(1024u << 16),
             (int)(64u << 16),
             KROWS,
             64,
             0, 0 };
  v4i gz = { 0, 0, 0, 0 };
#if __clang_major__ >= 23
  v8i gz8 = { 0, 0, 0, 0, 0, 0, 0, 0 };
  __builtin_amdgcn_tensor_load_to_lds(g0, g1, gz, gz, gz8, 0);
#else
  __builtin_amdgcn_tensor_load_to_lds(g0, g1, gz, gz, 0);
#endif
}
#endif

__global__ void attn_kernel(const float* __restrict__ qc,
                            const float* __restrict__ k,
                            const float* __restrict__ v,
                            float* __restrict__ wa2) {
  __shared__ __align__(16) float Qs[4 * HDIM];
  __shared__ float  Ssc[4][SKL];
  __shared__ float4 red4[256];
  __shared__ float  opart[4][4][HDIM];
#if USE_TDM
  __shared__ __align__(16) float Kbuf[2][KROWS * KPAD];
#endif

  int nh   = blockIdx.x;        // nseq*16 + h
  int nseq = nh >> 4, h = nh & 15;
  int t    = threadIdx.x;
  const float scale = 0.125f;   // 64^-0.5
  const float* kb = k + (size_t)nh * SKL * HDIM;
  const float* vb = v + (size_t)nh * SKL * HDIM;

  Qs[t] = qc[(size_t)nh * 256 + t];
  __syncthreads();
  const float4* Q4 = (const float4*)Qs;

  // ---- scores  S[b][j] = scale * (Q[b,:] . K[j,:])
#if USE_TDM
  {
    const int NT = SKL / KROWS;         // 16 tiles
    int wv = t >> 5;
    if (wv == 0 && t == 0) {            // descriptor issue: wave 0 only, one per workgroup
      tdm_load_k_tile(kb, (unsigned)(uintptr_t)&Kbuf[0][0], 0);
    }
    int r = t & 63, b = t >> 6;
    const float4* qb4 = Q4 + b * 16;
    for (int tile = 0; tile < NT; ++tile) {
      if (wv == 0 && t == 0 && tile + 1 < NT) {
        tdm_load_k_tile(kb, (unsigned)(uintptr_t)&Kbuf[(tile + 1) & 1][0],
                        tile + 1);
      }
      if (wv == 0) {
        if (tile + 1 < NT) __builtin_amdgcn_s_wait_tensorcnt(1);
        else               __builtin_amdgcn_s_wait_tensorcnt(0);
      }
      __syncthreads();                  // tile data visible to all waves
      const float4* kr = (const float4*)(&Kbuf[tile & 1][(size_t)r * KPAD]);
      float d = 0.f;
#pragma unroll
      for (int e = 0; e < 16; ++e) {
        float4 kk = kr[e];
        float4 qq = qb4[e];
        d += kk.x * qq.x + kk.y * qq.y + kk.z * qq.z + kk.w * qq.w;
      }
      Ssc[b][tile * KROWS + r] = d * scale;
      __syncthreads();                  // done reading buffer before reuse
    }
  }
#else
  for (int j = t; j < SKL; j += 256) {
    const float4* kr = (const float4*)(kb + (size_t)j * HDIM);
    __builtin_prefetch((const void*)(kb + (size_t)(j + 256) * HDIM), 0, 0);
    float d0 = 0.f, d1 = 0.f, d2 = 0.f, d3 = 0.f;
#pragma unroll
    for (int e = 0; e < 16; ++e) {
      float4 kk = kr[e];
      float4 q0 = Q4[e], q1 = Q4[16 + e], q2 = Q4[32 + e], q3 = Q4[48 + e];
      d0 += kk.x * q0.x + kk.y * q0.y + kk.z * q0.z + kk.w * q0.w;
      d1 += kk.x * q1.x + kk.y * q1.y + kk.z * q1.z + kk.w * q1.w;
      d2 += kk.x * q2.x + kk.y * q2.y + kk.z * q2.z + kk.w * q2.w;
      d3 += kk.x * q3.x + kk.y * q3.y + kk.z * q3.z + kk.w * q3.w;
    }
    Ssc[0][j] = d0 * scale; Ssc[1][j] = d1 * scale;
    Ssc[2][j] = d2 * scale; Ssc[3][j] = d3 * scale;
  }
  __syncthreads();
#endif

  // ---- row max (per batch b=0..3)
  float4 pm = make_float4(-1e30f, -1e30f, -1e30f, -1e30f);
  for (int j = t; j < SKL; j += 256) {
    pm.x = fmaxf(pm.x, Ssc[0][j]); pm.y = fmaxf(pm.y, Ssc[1][j]);
    pm.z = fmaxf(pm.z, Ssc[2][j]); pm.w = fmaxf(pm.w, Ssc[3][j]);
  }
  red4[t] = pm;
  __syncthreads();
  for (int o = 128; o > 0; o >>= 1) {
    if (t < o) {
      red4[t].x = fmaxf(red4[t].x, red4[t + o].x);
      red4[t].y = fmaxf(red4[t].y, red4[t + o].y);
      red4[t].z = fmaxf(red4[t].z, red4[t + o].z);
      red4[t].w = fmaxf(red4[t].w, red4[t + o].w);
    }
    __syncthreads();
  }
  float4 mx = red4[0];
  __syncthreads();

  // ---- exponentiate in place + row sums
  float4 ps = make_float4(0.f, 0.f, 0.f, 0.f);
  for (int j = t; j < SKL; j += 256) {
    float p0 = expf(Ssc[0][j] - mx.x); Ssc[0][j] = p0; ps.x += p0;
    float p1 = expf(Ssc[1][j] - mx.y); Ssc[1][j] = p1; ps.y += p1;
    float p2 = expf(Ssc[2][j] - mx.z); Ssc[2][j] = p2; ps.z += p2;
    float p3 = expf(Ssc[3][j] - mx.w); Ssc[3][j] = p3; ps.w += p3;
  }
  red4[t] = ps;
  __syncthreads();
  for (int o = 128; o > 0; o >>= 1) {
    if (t < o) {
      red4[t].x += red4[t + o].x; red4[t].y += red4[t + o].y;
      red4[t].z += red4[t + o].z; red4[t].w += red4[t + o].w;
    }
    __syncthreads();
  }
  float4 sm = red4[0];

  // ---- O = P·V : thread (quarter qr, hd) streams its quarter of V once,
  // fully coalesced across hd; P broadcast from LDS; prefetch ahead.
  int hd = t & 63, qr = t >> 6;
  float o0 = 0.f, o1 = 0.f, o2 = 0.f, o3 = 0.f;
  const float* vq = vb + (size_t)(qr * 256) * HDIM + hd;
  for (int j = 0; j < 256; ++j) {
    if ((j & 15) == 0)
      __builtin_prefetch((const void*)(vq + (size_t)(j + 32) * HDIM), 0, 0);
    float vv = vq[(size_t)j * HDIM];
    int jj = qr * 256 + j;
    o0 += Ssc[0][jj] * vv; o1 += Ssc[1][jj] * vv;
    o2 += Ssc[2][jj] * vv; o3 += Ssc[3][jj] * vv;
  }
  opart[qr][0][hd] = o0; opart[qr][1][hd] = o1;
  opart[qr][2][hd] = o2; opart[qr][3][hd] = o3;
  __syncthreads();

  int b = t >> 6;  // reuse t as (b, hd)
  float o = opart[0][b][hd] + opart[1][b][hd] + opart[2][b][hd] + opart[3][b][hd];
  float smb = (b == 0) ? sm.x : (b == 1) ? sm.y : (b == 2) ? sm.z : sm.w;
  wa2[((size_t)b * 1024 + nseq) * DIM + h * HDIM + hd] = o / smb;
}

// ---------------------------------------------------------------------------
extern "C" void kernel_launch(void* const* d_in, const int* in_sizes, int n_in,
                              void* d_out, int out_size, void* d_ws, size_t ws_size,
                              hipStream_t stream) {
  (void)in_sizes; (void)n_in; (void)out_size; (void)ws_size;
  const float* x         = (const float*)d_in[0];
  const float* k         = (const float*)d_in[1];
  const float* v         = (const float*)d_in[2];
  const float* qkv_w     = (const float*)d_in[3];
  const float* qkv_b     = (const float*)d_in[4];
  const float* sa_proj_w = (const float*)d_in[5];
  const float* sa_proj_b = (const float*)d_in[6];
  const float* q_w       = (const float*)d_in[7];
  const float* q_b       = (const float*)d_in[8];
  const float* ed_proj_w = (const float*)d_in[9];
  const float* ed_proj_b = (const float*)d_in[10];
  const float* fc1_w     = (const float*)d_in[11];
  const float* fc1_b     = (const float*)d_in[12];
  const float* fc2_w     = (const float*)d_in[13];
  const float* fc2_b     = (const float*)d_in[14];
  const float* ln1_w     = (const float*)d_in[15];
  const float* ln1_b     = (const float*)d_in[16];
  const float* ln2_w     = (const float*)d_in[17];
  const float* ln2_b     = (const float*)d_in[18];
  const float* ln3_w     = (const float*)d_in[19];
  const float* ln3_b     = (const float*)d_in[20];

  float* wsf   = (float*)d_ws;
  float* xsum  = wsf;                 // 1024
  float* vred  = wsf + 1024;          // 64
  float* sarow = wsf + 2048;          // 1024
  float* big   = wsf + 4096;
  const size_t SLOT = (size_t)4 * 1024 * 1024;   // 4M floats = 16 MB
  float* S0 = big;                    // h1, later ca
  float* S1 = big + SLOT;             // qc, later h2
  float* S2 = big + 2 * SLOT;         // wa2, later z2
  float* S3 = big + 3 * SLOT;         // z1 (4096x4096 = 16M floats)

  // 1) collapsed self-attention row
  colsum_kernel<<<4, 256, 0, stream>>>(x, xsum);
  vred_kernel<<<1, 64, 0, stream>>>(xsum, qkv_w, qkv_b, vred);
  sarow_kernel<<<4, 256, 0, stream>>>(vred, sa_proj_w, sa_proj_b, sarow);
  // 2) h1 = LN(x + sa_row)
  add_ln_kernel<<<BN, 256, 0, stream>>>(x, sarow, 1, ln1_w, ln1_b, S0);
  // 3) qc = h1 @ q_w + q_b  ->  [N,H,B,HD]
  qc_gemm_kernel<<<2048, 256, 0, stream>>>(S0, q_w, q_b, S1);
  // 4) enc-dec attention (streams 8 GB of K/V; K via TDM)
  attn_kernel<<<16384, 256, 0, stream>>>(S1, k, v, S2);
  // 5) ca = wa2 @ ed_proj_w + b ; h2 = LN(ca + x)
  gemm_bias_kernel<<<2048, 256, 0, stream>>>(S2, ed_proj_w, ed_proj_b, S0,
                                             BN, DIM, DIM, 0);
  add_ln_kernel<<<BN, 256, 0, stream>>>(S0, x, 0, ln2_w, ln2_b, S1);
  // 6) MLP: z1 = gelu(h2 @ fc1 + b); z2 = z1 @ fc2 + b; out = LN(z2 + h2)
  gemm_bias_kernel<<<8192, 256, 0, stream>>>(S1, fc1_w, fc1_b, S3,
                                             BN, 4 * DIM, DIM, 1);
  gemm_bias_kernel<<<2048, 256, 0, stream>>>(S3, fc2_w, fc2_b, S2,
                                             BN, DIM, 4 * DIM, 0);
  add_ln_kernel<<<BN, 256, 0, stream>>>(S2, S1, 0, ln3_w, ln3_b, (float*)d_out);
}
